// StaticSceneTwoStreamClassifier_86406152061387
// MI455X (gfx1250) — matrix operand order repaired and checked by reading, not compiled, not run on hardware
//
#include <hip/hip_runtime.h>

#define NPTS   32768
#define PPTS   2048
#define BATCH  16
#define KNEIGH 20
#define EDGES  (NPTS * KNEIGH)
#define EPSV   1e-5f

typedef __attribute__((ext_vector_type(16))) __bf16 v16bf;
typedef __attribute__((ext_vector_type(8)))  float  v8f;

union ABfrag { v16bf v; __bf16 e[16]; };

// ---------------- input BN ----------------
__global__ void in_stats_kernel(const float* __restrict__ x, float* __restrict__ st) {
    __shared__ float sh[8];
    int tid = threadIdx.x;
    if (tid < 8) sh[tid] = 0.f;
    __syncthreads();
    float ls[4] = {0,0,0,0}, lq[4] = {0,0,0,0};
    for (int r = blockIdx.x * blockDim.x + tid; r < NPTS; r += gridDim.x * blockDim.x) {
        #pragma unroll
        for (int c = 0; c < 4; c++) { float v = x[r*4+c]; ls[c] += v; lq[c] += v*v; }
    }
    #pragma unroll
    for (int c = 0; c < 4; c++) { atomicAdd(&sh[c], ls[c]); atomicAdd(&sh[4+c], lq[c]); }
    __syncthreads();
    if (tid < 8) atomicAdd(&st[tid], sh[tid]);
}

__global__ void finalize_input_kernel(const float* st, const float* sg, const float* sb,
                                      const float* tg, const float* tb,
                                      float* scale, float* shift) {
    int c = threadIdx.x; if (c >= 4) return;
    float mean = st[c] / (float)NPTS;
    float var  = st[4+c] / (float)NPTS - mean*mean;
    float g = (c < 3) ? sg[c] : tg[0];
    float b = (c < 3) ? sb[c] : tb[0];
    float sc = g * rsqrtf(var + EPSV);
    scale[c] = sc; shift[c] = b - mean*sc;
}

__global__ void normalize_kernel(const float* __restrict__ x, const float* scale,
                                 const float* shift, float* __restrict__ pers) {
    int i = blockIdx.x * blockDim.x + threadIdx.x;
    if (i >= NPTS * 4) return;
    int c = i & 3;
    pers[i] = x[i] * scale[c] + shift[c];
}

// ---------------- weight conversion (f32 -> padded bf16, row-major K x N) -----
__global__ void convw_kernel(const float* __restrict__ W, int din, int Hout, int Kpad,
                             __bf16* __restrict__ out) {
    int i = blockIdx.x * blockDim.x + threadIdx.x;
    if (i >= Kpad * Hout) return;
    int k = i / Hout;
    out[i] = (k < din) ? (__bf16)W[i] : (__bf16)0.f;
}

// ---------------- brute-force KNN (top-K=20 smallest d2, excl self) ----------
template<int D, int S>
__global__ void knn_kernel(const float* __restrict__ feat, int* __restrict__ idxout) {
    int t = blockIdx.x * blockDim.x + threadIdx.x;
    if (t >= NPTS) return;
    int b = t / PPTS, p = t % PPTS;
    const float* base = feat + (size_t)b * PPTS * S;
    float q[D];
    #pragma unroll
    for (int i = 0; i < D; i++) q[i] = base[(size_t)p * S + i];
    float bd[KNEIGH]; int bi[KNEIGH];
    #pragma unroll
    for (int i = 0; i < KNEIGH; i++) { bd[i] = 3.0e38f; bi[i] = 0; }
    for (int j = 0; j < PPTS; j++) {
        if (j == p) continue;
        const float* r = base + (size_t)j * S;
        float d2 = 0.f;
        #pragma unroll
        for (int i = 0; i < D; i++) { float df = q[i] - r[i]; d2 += df * df; }
        if (d2 < bd[KNEIGH-1]) {
            int pos = KNEIGH - 1;
            while (pos > 0 && bd[pos-1] > d2) { bd[pos] = bd[pos-1]; bi[pos] = bi[pos-1]; pos--; }
            bd[pos] = d2; bi[pos] = j;
        }
    }
    for (int k = 0; k < KNEIGH; k++) idxout[(size_t)t * KNEIGH + k] = b * PPTS + bi[k];
}

// ---------------- edge layer-1 WMMA GEMM: e=[xi, xj-xi] @ W1 + b1 ------------
// D = point feature dim (edge dim = 2D), HOUT = output channels.
template<int D, int HOUT>
__global__ void edge_gemm1_kernel(const float* __restrict__ feat, int fstride,
                                  const int* __restrict__ idx,
                                  const __bf16* __restrict__ Wb,
                                  const float* __restrict__ bias,
                                  float* __restrict__ out,
                                  float* __restrict__ gsum, float* __restrict__ gsq,
                                  int mtiles, int chunk) {
    constexpr int DD2 = 2 * D;
    constexpr int KT  = (DD2 + 31) / 32;
    constexpr int NT  = HOUT / 16;
    int wid  = (blockIdx.x * blockDim.x + threadIdx.x) >> 5;
    int lane = threadIdx.x & 31;
    int mgroups = (mtiles + chunk - 1) / chunk;
    if (wid >= NT * mgroups) return;
    int ntile = wid % NT;
    int mg    = wid / NT;
    int mstart = mg * chunk;
    int mend = mstart + chunk; if (mend > mtiles) mend = mtiles;
    int m16 = lane & 15, hf = lane >> 4;
    int col = ntile * 16 + m16;

    ABfrag Bf[KT];
    #pragma unroll
    for (int kt = 0; kt < KT; ++kt) {
        #pragma unroll
        for (int j = 0; j < 8; j++) {
            int k = kt * 32 + hf * 16 + 2 * j;
            Bf[kt].e[2*j]   = Wb[(size_t)k * HOUT + col];
            Bf[kt].e[2*j+1] = Wb[(size_t)(k+1) * HOUT + col];
        }
    }
    float bv = bias[col];
    float wsum = 0.f, wsq = 0.f;
    for (int mt = mstart; mt < mend; ++mt) {
        int row = mt * 16 + m16;
        int n  = row / KNEIGH;
        int jr = idx[row];
        const float* __restrict__ Fi = feat + (size_t)n  * fstride;
        const float* __restrict__ Fj = feat + (size_t)jr * fstride;
        v8f acc;
        #pragma unroll
        for (int r = 0; r < 8; r++) acc[r] = 0.f;
        #pragma unroll
        for (int kt = 0; kt < KT; ++kt) {
            ABfrag A;
            #pragma unroll
            for (int j = 0; j < 8; j++) {
                int k = kt * 32 + hf * 16 + 2 * j;
                float v0 = (k   < D) ? Fi[k]  : ((k   < DD2) ? (Fj[k-D]  - Fi[k-D])  : 0.f);
                int k1 = k + 1;
                float v1 = (k1  < D) ? Fi[k1] : ((k1  < DD2) ? (Fj[k1-D] - Fi[k1-D]) : 0.f);
                A.e[2*j]   = (__bf16)v0;
                A.e[2*j+1] = (__bf16)v1;
            }
            acc = __builtin_amdgcn_wmma_f32_16x16x32_bf16(false, A.v, false, Bf[kt].v,
                                                          (short)0, acc, false, false);
        }
        size_t rb = (size_t)(mt * 16 + hf * 8);
        #pragma unroll
        for (int r = 0; r < 8; r++) {
            float v = acc[r] + bv;
            out[(rb + r) * (size_t)HOUT + col] = v;
            wsum += v; wsq += v * v;
        }
    }
    wsum += __shfl_xor(wsum, 16, 32);
    wsq  += __shfl_xor(wsq,  16, 32);
    if (lane < 16) { atomicAdd(&gsum[col], wsum); atomicAdd(&gsq[col], wsq); }
}

// ---------------- generic WMMA GEMM: [relu(bn(A))] @ W + b -------------------
template<int HIN, int HOUT, bool BN>
__global__ void gemm2_kernel(const float* __restrict__ Abuf,
                             const float* __restrict__ scIn, const float* __restrict__ shIn,
                             const __bf16* __restrict__ Wb,
                             const float* __restrict__ bias,
                             float* __restrict__ out,
                             float* __restrict__ gsum, float* __restrict__ gsq,
                             int mtiles, int chunk) {
    constexpr int KT = HIN / 32;
    constexpr int NT = HOUT / 16;
    int wid  = (blockIdx.x * blockDim.x + threadIdx.x) >> 5;
    int lane = threadIdx.x & 31;
    int mgroups = (mtiles + chunk - 1) / chunk;
    if (wid >= NT * mgroups) return;
    int ntile = wid % NT;
    int mg    = wid / NT;
    int mstart = mg * chunk;
    int mend = mstart + chunk; if (mend > mtiles) mend = mtiles;
    int m16 = lane & 15, hf = lane >> 4;
    int col = ntile * 16 + m16;

    ABfrag Bf[KT];
    #pragma unroll
    for (int kt = 0; kt < KT; ++kt) {
        #pragma unroll
        for (int j = 0; j < 8; j++) {
            int k = kt * 32 + hf * 16 + 2 * j;
            Bf[kt].e[2*j]   = Wb[(size_t)k * HOUT + col];
            Bf[kt].e[2*j+1] = Wb[(size_t)(k+1) * HOUT + col];
        }
    }
    // Hoist BN scale/shift for this lane's fixed K positions.
    float sck[KT * 16], shk[KT * 16];
    if (BN) {
        #pragma unroll
        for (int kt = 0; kt < KT; ++kt) {
            #pragma unroll
            for (int j = 0; j < 8; j++) {
                int k = kt * 32 + hf * 16 + 2 * j;
                sck[kt*16 + 2*j]   = scIn[k];
                sck[kt*16 + 2*j+1] = scIn[k+1];
                shk[kt*16 + 2*j]   = shIn[k];
                shk[kt*16 + 2*j+1] = shIn[k+1];
            }
        }
    }
    float bv = bias[col];
    float wsum = 0.f, wsq = 0.f;
    for (int mt = mstart; mt < mend; ++mt) {
        int row = mt * 16 + m16;
        const float* __restrict__ Ar = Abuf + (size_t)row * HIN;
        v8f acc;
        #pragma unroll
        for (int r = 0; r < 8; r++) acc[r] = 0.f;
        #pragma unroll
        for (int kt = 0; kt < KT; ++kt) {
            ABfrag A;
            #pragma unroll
            for (int j = 0; j < 8; j++) {
                int k = kt * 32 + hf * 16 + 2 * j;
                float v0 = Ar[k], v1 = Ar[k+1];
                if (BN) {
                    v0 = fmaxf(v0 * sck[kt*16 + 2*j]   + shk[kt*16 + 2*j],   0.f);
                    v1 = fmaxf(v1 * sck[kt*16 + 2*j+1] + shk[kt*16 + 2*j+1], 0.f);
                }
                A.e[2*j]   = (__bf16)v0;
                A.e[2*j+1] = (__bf16)v1;
            }
            acc = __builtin_amdgcn_wmma_f32_16x16x32_bf16(false, A.v, false, Bf[kt].v,
                                                          (short)0, acc, false, false);
        }
        size_t rb = (size_t)(mt * 16 + hf * 8);
        #pragma unroll
        for (int r = 0; r < 8; r++) {
            float v = acc[r] + bv;
            out[(rb + r) * (size_t)HOUT + col] = v;
            wsum += v; wsq += v * v;
        }
    }
    wsum += __shfl_xor(wsum, 16, 32);
    wsq  += __shfl_xor(wsq,  16, 32);
    if (lane < 16) { atomicAdd(&gsum[col], wsum); atomicAdd(&gsq[col], wsq); }
}

// ---------------- BN affine finalize -----------------------------------------
__global__ void finalize_affine_kernel(const float* gsum, const float* gsq,
                                       const float* g, const float* b, float count,
                                       float* scale, float* shift, int C) {
    int c = blockIdx.x * blockDim.x + threadIdx.x; if (c >= C) return;
    float mean = gsum[c] / count;
    float var  = gsq[c] / count - mean * mean;
    float sc = g[c] * rsqrtf(var + EPSV);
    scale[c] = sc; shift[c] = b[c] - mean * sc;
}

// ---------------- max over K of relu(bn(pre2)) -> feats[:, colofs:+H] --------
__global__ void maxpool_kernel(const float* __restrict__ pre, int Hc,
                               const float* __restrict__ scale, const float* __restrict__ shift,
                               float* __restrict__ feats, int colofs) {
    int t = blockIdx.x * blockDim.x + threadIdx.x;
    if (t >= NPTS * Hc) return;
    int n = t / Hc, c = t % Hc;
    float sc = scale[c], sh = shift[c], m = 0.f;
    size_t base = (size_t)n * KNEIGH * Hc + c;
    for (int k = 0; k < KNEIGH; k++) {
        float v = pre[base + (size_t)k * Hc] * sc + sh;
        v = fmaxf(v, 0.f);
        m = fmaxf(m, v);
    }
    feats[(size_t)n * 128 + colofs + c] = m;
}

// ---------------- classifier tail: relu(bn(preC)) @ Wc2 + bc2 ----------------
__global__ void cls_final_kernel(const float* __restrict__ pre, const float* scale,
                                 const float* shift, const float* __restrict__ Wc2,
                                 const float* bc2, float* __restrict__ out) {
    int r = blockIdx.x * blockDim.x + threadIdx.x;
    if (r >= NPTS) return;
    float acc = 0.f;
    #pragma unroll 8
    for (int c = 0; c < 128; c++) {
        float h = pre[(size_t)r * 128 + c] * scale[c] + shift[c];
        h = fmaxf(h, 0.f);
        acc += h * Wc2[c];
    }
    out[r] = acc + bc2[0];
}

// ---------------- host orchestration -----------------------------------------
template<int D, int HOUT>
static void run_conv(const float* feat, int fstride,
                     const float* const* P,  // W1,b1,g1,be1,W2,b2,g2,be2
                     int colofs,
                     int* idxb, float* buf1, float* buf2, __bf16* wbf,
                     float* gsum, float* gsq, float* scA, float* shA, float* scB, float* shB,
                     float* feats, hipStream_t stream) {
    const int CHUNK = 64;
    const int mtiles = EDGES / 16;
    constexpr int NT = HOUT / 16;
    constexpr int DD2 = 2 * D;
    constexpr int Kpad1 = ((DD2 + 31) / 32) * 32;

    convw_kernel<<<(Kpad1 * HOUT + 255) / 256, 256, 0, stream>>>(P[0], DD2, HOUT, Kpad1, wbf);
    hipMemsetAsync(gsum, 0, 128 * sizeof(float), stream);
    hipMemsetAsync(gsq,  0, 128 * sizeof(float), stream);
    int mgroups = (mtiles + CHUNK - 1) / CHUNK;
    int thr = NT * mgroups * 32;
    edge_gemm1_kernel<D, HOUT><<<(thr + 127) / 128, 128, 0, stream>>>(
        feat, fstride, idxb, wbf, P[1], buf1, gsum, gsq, mtiles, CHUNK);
    finalize_affine_kernel<<<1, HOUT, 0, stream>>>(gsum, gsq, P[2], P[3], (float)EDGES, scA, shA, HOUT);

    convw_kernel<<<(HOUT * HOUT + 255) / 256, 256, 0, stream>>>(P[4], HOUT, HOUT, HOUT, wbf);
    hipMemsetAsync(gsum, 0, 128 * sizeof(float), stream);
    hipMemsetAsync(gsq,  0, 128 * sizeof(float), stream);
    gemm2_kernel<HOUT, HOUT, true><<<(thr + 127) / 128, 128, 0, stream>>>(
        buf1, scA, shA, wbf, P[5], buf2, gsum, gsq, mtiles, CHUNK);
    finalize_affine_kernel<<<1, HOUT, 0, stream>>>(gsum, gsq, P[6], P[7], (float)EDGES, scB, shB, HOUT);

    maxpool_kernel<<<(NPTS * HOUT + 255) / 256, 256, 0, stream>>>(buf2, HOUT, scB, shB, feats, colofs);
}

extern "C" void kernel_launch(void* const* d_in, const int* in_sizes, int n_in,
                              void* d_out, int out_size, void* d_ws, size_t ws_size,
                              hipStream_t stream) {
    (void)in_sizes; (void)n_in; (void)out_size; (void)ws_size;
    const float* x   = (const float*)d_in[0];
    const float* sg  = (const float*)d_in[1];
    const float* sb  = (const float*)d_in[2];
    const float* tg  = (const float*)d_in[3];
    const float* tbp = (const float*)d_in[4];
    const float* Ps[3][8];
    for (int c = 0; c < 3; c++)
        for (int j = 0; j < 8; j++)
            Ps[c][j] = (const float*)d_in[5 + c * 8 + j];
    const float* Wc1  = (const float*)d_in[29];
    const float* bc1  = (const float*)d_in[30];
    const float* gc1  = (const float*)d_in[31];
    const float* bec1 = (const float*)d_in[32];
    const float* Wc2  = (const float*)d_in[33];
    const float* bc2  = (const float*)d_in[34];

    float* ws = (float*)d_ws;
    size_t off = 0;
    float* pers  = ws + off; off += (size_t)NPTS * 4;
    float* feats = ws + off; off += (size_t)NPTS * 128;
    int*   idxb  = (int*)(ws + off); off += (size_t)EDGES;
    float* buf1  = ws + off; off += (size_t)EDGES * 64;
    float* buf2  = ws + off; off += (size_t)EDGES * 64;
    __bf16* wbf  = (__bf16*)(ws + off); off += 8192; // 16384 bf16
    float* gsum = ws + off; off += 128;
    float* gsq  = ws + off; off += 128;
    float* scA  = ws + off; off += 128;
    float* shA  = ws + off; off += 128;
    float* scB  = ws + off; off += 128;
    float* shB  = ws + off; off += 128;
    float* st8  = ws + off; off += 8;

    // input BN -> pers (N x 4): cols 0..2 spatial, col 3 time
    hipMemsetAsync(st8, 0, 8 * sizeof(float), stream);
    in_stats_kernel<<<64, 256, 0, stream>>>(x, st8);
    finalize_input_kernel<<<1, 4, 0, stream>>>(st8, sg, sb, tg, tbp, scA, shA);
    normalize_kernel<<<(NPTS * 4 + 255) / 256, 256, 0, stream>>>(x, scA, shA, pers);

    // conv_spatial: knn on xs (d=3, stride 4), H=32 -> feats[:,0:32]
    knn_kernel<3, 4><<<(NPTS + 127) / 128, 128, 0, stream>>>(pers, idxb);
    run_conv<3, 32>(pers, 4, Ps[0], 0, idxb, buf1, buf2, wbf,
                    gsum, gsq, scA, shA, scB, shB, feats, stream);

    // conv_persistence: knn on pers (d=4, stride 4), H=32 -> feats[:,32:64]
    knn_kernel<4, 4><<<(NPTS + 127) / 128, 128, 0, stream>>>(pers, idxb);
    run_conv<4, 32>(pers, 4, Ps[1], 32, idxb, buf1, buf2, wbf,
                    gsum, gsq, scA, shA, scB, shB, feats, stream);

    // conv_fusion: knn on comb = feats[:,0:64] (d=64, stride 128), H=64 -> feats[:,64:128]
    knn_kernel<64, 128><<<(NPTS + 127) / 128, 128, 0, stream>>>(feats, idxb);
    run_conv<64, 64>(feats, 128, Ps[2], 64, idxb, buf1, buf2, wbf,
                     gsum, gsq, scA, shA, scB, shB, feats, stream);

    // classifier: feats(N x 128) @ Wc1(128x128) -> buf1, BN, relu, @ Wc2 -> out
    convw_kernel<<<(128 * 128 + 255) / 256, 256, 0, stream>>>(Wc1, 128, 128, 128, wbf);
    hipMemsetAsync(gsum, 0, 128 * sizeof(float), stream);
    hipMemsetAsync(gsq,  0, 128 * sizeof(float), stream);
    {
        const int CHUNK = 64;
        int mtiles = NPTS / 16;
        int mgroups = (mtiles + CHUNK - 1) / CHUNK;
        int thr = 8 * mgroups * 32;
        gemm2_kernel<128, 128, false><<<(thr + 127) / 128, 128, 0, stream>>>(
            feats, nullptr, nullptr, wbf, bc1, buf1, gsum, gsq, mtiles, CHUNK);
    }
    finalize_affine_kernel<<<1, 128, 0, stream>>>(gsum, gsq, gc1, bec1, (float)NPTS, scA, shA, 128);
    cls_final_kernel<<<(NPTS + 255) / 256, 256, 0, stream>>>(buf1, scA, shA, Wc2, bc2, (float*)d_out);
}
